// SelfAttention_33346126086785
// MI455X (gfx1250) — matrix operand order repaired and checked by reading, not compiled, not run on hardware
//
#include <hip/hip_runtime.h>
#include <hip/hip_bf16.h>
#include <math.h>

typedef __attribute__((ext_vector_type(16))) __bf16 v16bf;
typedef __attribute__((ext_vector_type(8)))  __bf16 v8bf;
typedef __attribute__((ext_vector_type(8)))  float  v8f;

#define B_   2
#define N_   2048
#define E_   1024
#define H_   16
#define D_   64
#define ROWS (B_ * N_)   // 4096
#define F3   (3 * E_)    // 3072
#define BH   (B_ * H_)   // 32

#define USE_ASYNC 1

__device__ inline v8f wmma_bf16(v16bf a, v16bf b, v8f c) {
  return __builtin_amdgcn_wmma_f32_16x16x32_bf16(false, a, false, b, (short)0, c, false, false);
}

__device__ inline v16bf cat8(v8bf lo, v8bf hi) {
  return __builtin_shufflevector(lo, hi, 0, 1, 2, 3, 4, 5, 6, 7,
                                         8, 9, 10, 11, 12, 13, 14, 15);
}

#if USE_ASYNC
__device__ inline unsigned lds_addr(const void* p) {
  return (unsigned)(size_t)(__attribute__((address_space(3))) const void*)p;
}
// GLOBAL_LOAD_ASYNC_TO_LDS_B128 (VGLOBAL op 98): VDST = LDS byte address,
// VADDR = 64-bit global address, GV mode (SADDR = off).  Tracked by ASYNCcnt.
__device__ inline void async_ld_b128(unsigned ldsoff, const void* gptr) {
  asm volatile("global_load_async_to_lds_b128 %0, %1, off"
               :: "v"(ldsoff), "v"(gptr) : "memory");
}
__device__ inline void wait_async0() {
  asm volatile("s_wait_asynccnt 0x0" ::: "memory");
}
#endif

// ---------------------------------------------------------------------------
// Kernel 1: fused LayerNorm -> bf16, one block per row of x [4096 x 1024]
// ---------------------------------------------------------------------------
__global__ void __launch_bounds__(256) ln_norm_kernel(
    const float* __restrict__ x, const float* __restrict__ gamma,
    const float* __restrict__ beta, __bf16* __restrict__ xn) {
  const int row = blockIdx.x;
  const float* xr = x + (size_t)row * E_;
  float v[4], s = 0.f, s2 = 0.f;
  #pragma unroll
  for (int i = 0; i < 4; ++i) {
    v[i] = xr[threadIdx.x + 256 * i];
    s += v[i]; s2 += v[i] * v[i];
  }
  __shared__ float sh0[256], sh1[256];
  __shared__ float mu_s, rstd_s;
  sh0[threadIdx.x] = s; sh1[threadIdx.x] = s2;
  __syncthreads();
  for (int off = 128; off > 0; off >>= 1) {
    if (threadIdx.x < off) {
      sh0[threadIdx.x] += sh0[threadIdx.x + off];
      sh1[threadIdx.x] += sh1[threadIdx.x + off];
    }
    __syncthreads();
  }
  if (threadIdx.x == 0) {
    float mu  = sh0[0] * (1.0f / E_);
    float var = sh1[0] * (1.0f / E_) - mu * mu;
    mu_s = mu; rstd_s = rsqrtf(var + 1e-5f);
  }
  __syncthreads();
  const float mu = mu_s, rstd = rstd_s;
  #pragma unroll
  for (int i = 0; i < 4; ++i) {
    const int idx = threadIdx.x + 256 * i;
    xn[(size_t)row * E_ + idx] =
        (__bf16)((v[i] - mu) * rstd * gamma[idx] + beta[idx]);
  }
}

// ---------------------------------------------------------------------------
// Kernel 2: convert w_qkv fp32 -> bf16 (reused 128x by the GEMM)
// ---------------------------------------------------------------------------
__global__ void __launch_bounds__(256) cvt_w_kernel(const float* __restrict__ w,
                                                    __bf16* __restrict__ wb) {
  const int base = (blockIdx.x * 256 + threadIdx.x) * 4;
  #pragma unroll
  for (int i = 0; i < 4; ++i) wb[base + i] = (__bf16)w[base + i];
}

// ---------------------------------------------------------------------------
// Kernel 3: QKV projection, WMMA bf16.  Wave computes a 32x64 tile
// (8 WMMAs / 32-k step vs 6 fragment loads).  64-wide N tile = one
// (matrix, head).  Q pre-scaled by D^-0.5; V stored TRANSPOSED
// (Vt[b,h][d][n]) so attention's P*V B-operand is contiguous.
// ---------------------------------------------------------------------------
__global__ void __launch_bounds__(256) qkv_gemm_kernel(
    const __bf16* __restrict__ xn, const __bf16* __restrict__ wb,
    __bf16* __restrict__ heads) {
  const int lane = threadIdx.x & 31;
  const int wave = threadIdx.x >> 5;
  const int tile = blockIdx.x * 8 + wave;
  const int NT   = F3 / 64;          // 48
  const int nT   = tile % NT;
  const int mT   = tile / NT;        // 0..127, 32-row tiles
  const int n    = lane & 15;
  const int hh   = lane >> 4;

  const __bf16* a0r = xn + (size_t)(mT * 32 + n) * E_;
  const __bf16* a1r = xn + (size_t)(mT * 32 + 16 + n) * E_;

  v8f acc[2][4] = {};

  for (int kb = 0; kb < E_; kb += 32) {
    // A fragments: contiguous chunk [kb+16h .. +15]
    v16bf a0 = *(const v16bf*)(a0r + kb + 16 * hh);
    v16bf a1 = *(const v16bf*)(a1r + kb + 16 * hh);
    // B fragments: column f = nT*64 + 16t + n, K chunk kb + 16h (32B)
    const int kB = kb + 16 * hh;
    #pragma unroll
    for (int t = 0; t < 4; ++t) {
      v16bf b = *(const v16bf*)(wb + (size_t)(nT * 64 + 16 * t + n) * E_ + kB);
      acc[0][t] = wmma_bf16(a0, b, acc[0][t]);
      acc[1][t] = wmma_bf16(a1, b, acc[1][t]);
    }
  }

  const int   mat   = nT / 16;   // 0=Q, 1=K, 2=V
  const int   head  = nT % 16;
  const float scale = (mat == 0) ? 0.125f : 1.0f;
  #pragma unroll
  for (int s2 = 0; s2 < 2; ++s2) {
    #pragma unroll
    for (int j = 0; j < 8; ++j) {
      const int r  = mT * 32 + 16 * s2 + j + 8 * hh;
      const int bb = r / N_;
      const int nr = r % N_;
      const int bh = bb * H_ + head;
      if (mat == 2) {
        // transposed store: Vt[bh][d][nr], d = 16t + n
        __bf16* dst = heads + (size_t)(2 * BH + bh) * N_ * D_ + nr;
        #pragma unroll
        for (int t = 0; t < 4; ++t)
          dst[(size_t)(16 * t + n) * N_] = (__bf16)(acc[s2][t][j] * scale);
      } else {
        __bf16* dst = heads + ((size_t)(mat * BH + bh) * N_ + nr) * D_;
        #pragma unroll
        for (int t = 0; t < 4; ++t)
          dst[16 * t + n] = (__bf16)(acc[s2][t][j] * scale);
      }
    }
  }
}

// ---------------------------------------------------------------------------
// Kernel 4: causal flash attention.  Block = 8 waves = 8 consecutive
// 16-query tiles of one (b,h).  K/Vt 32-key tiles staged cooperatively in
// LDS via async-to-LDS (ASYNCcnt) and shared by all waves.
// ---------------------------------------------------------------------------
__global__ void __launch_bounds__(256) attn_kernel(const __bf16* __restrict__ heads,
                                                   float* __restrict__ out) {
  const int lane = threadIdx.x & 31;
  const int wave = threadIdx.x >> 5;
  const int blocksPerBH = (N_ / 16) / 8;              // 16
  const int bh     = blockIdx.x / blocksPerBH;        // uniform per block
  const int qTb    = (blockIdx.x % blocksPerBH) * 8;
  const int qBase  = (qTb + wave) * 16;
  const int blockQ = qTb * 16;

  const __bf16* Q  = heads + (size_t)(0 * BH + bh) * N_ * D_;
  const __bf16* K  = heads + (size_t)(1 * BH + bh) * N_ * D_;
  const __bf16* Vt = heads + (size_t)(2 * BH + bh) * N_ * D_;  // [D][N]

  const int n  = lane & 15;
  const int hh = lane >> 4;

  // Q A-fragments for the two 32-wide d k-steps (Q pre-scaled).
  v16bf aq0, aq1;
  {
    const __bf16* qr = Q + (size_t)(qBase + n) * D_;
    aq0 = *(const v16bf*)(qr + 16 * hh);
    aq1 = *(const v16bf*)(qr + 32 + 16 * hh);
  }

  float mrow[8], lrow[8];
  v8f o0 = {}, o1 = {}, o2 = {}, o3 = {};
  #pragma unroll
  for (int j = 0; j < 8; ++j) { mrow[j] = -1e30f; lrow[j] = 0.f; }

  __shared__ __align__(32) __bf16 Kt[32][64];      // [key][d]
  __shared__ __align__(32) __bf16 Vs[64][32];      // [d][key]
  __shared__ __align__(32) __bf16 Pl[8][16][32];   // per-wave P transpose tile
  __bf16 (*P)[32] = Pl[wave];

  const int nb_w  = qBase / 32 + 1;       // this wave's causal block count
  const int nbmax = blockQ / 32 + 4;      // block-wide bound

  for (int kb = 0; kb < nbmax; ++kb) {
    const int kB = kb * 32;
    // ---- cooperative staging of K tile (4KB) + Vt tile (4KB) ----
    {
      const int key = threadIdx.x >> 3, ck = (threadIdx.x & 7) * 8;   // K
      const int d   = threadIdx.x >> 2, cv = (threadIdx.x & 3) * 8;   // Vt
      const __bf16* kg = K  + (size_t)(kB + key) * D_ + ck;
      const __bf16* vg = Vt + (size_t)d * N_ + kB + cv;
#if USE_ASYNC
      async_ld_b128(lds_addr(&Kt[key][ck]), kg);
      async_ld_b128(lds_addr(&Vs[d][cv]), vg);
      wait_async0();
#else
      *(v8bf*)&Kt[key][ck] = *(const v8bf*)kg;
      *(v8bf*)&Vs[d][cv]   = *(const v8bf*)vg;
#endif
      if (kb + 1 < nbmax)
        __builtin_prefetch(K + (size_t)(kB + 32 + key) * D_, 0, 0);
    }
    __syncthreads();

    if (kb < nb_w) {
      // ---- S = Q K^T : two 16-key fragments, K-dim 64 (2 WMMAs each) ----
      v8f s0 = {}, s1 = {};
      {
        v16bf bk;
        bk = *(const v16bf*)&Kt[n][16 * hh];            s0 = wmma_bf16(aq0, bk, s0);
        bk = *(const v16bf*)&Kt[n][32 + 16 * hh];       s0 = wmma_bf16(aq1, bk, s0);
        bk = *(const v16bf*)&Kt[16 + n][16 * hh];       s1 = wmma_bf16(aq0, bk, s1);
        bk = *(const v16bf*)&Kt[16 + n][32 + 16 * hh];  s1 = wmma_bf16(aq1, bk, s1);
      }

      // ---- causal mask + online softmax ----
      float al[8];
      #pragma unroll
      for (int j = 0; j < 8; ++j) {
        const int q = qBase + j + 8 * hh;
        if (kB + n > q)      s0[j] = -1e30f;
        if (kB + 16 + n > q) s1[j] = -1e30f;
        float mx = fmaxf(s0[j], s1[j]);
        mx = fmaxf(mx, __shfl_xor(mx, 1));
        mx = fmaxf(mx, __shfl_xor(mx, 2));
        mx = fmaxf(mx, __shfl_xor(mx, 4));
        mx = fmaxf(mx, __shfl_xor(mx, 8));
        const float mnew  = fmaxf(mrow[j], mx);
        const float alpha = __expf(mrow[j] - mnew);
        mrow[j] = mnew;
        const float p0 = __expf(s0[j] - mnew);
        const float p1 = __expf(s1[j] - mnew);
        s0[j] = p0; s1[j] = p1;
        float rs = p0 + p1;
        rs += __shfl_xor(rs, 1);
        rs += __shfl_xor(rs, 2);
        rs += __shfl_xor(rs, 4);
        rs += __shfl_xor(rs, 8);
        lrow[j] = lrow[j] * alpha + rs;
        al[j]   = alpha;
      }
      #pragma unroll
      for (int j = 0; j < 8; ++j) {
        o0[j] *= al[j]; o1[j] *= al[j]; o2[j] *= al[j]; o3[j] *= al[j];
      }

      // ---- transpose P through per-wave LDS tile: C-layout -> A-layout ----
      #pragma unroll
      for (int j = 0; j < 8; ++j) {
        P[j + 8 * hh][n]      = (__bf16)s0[j];
        P[j + 8 * hh][16 + n] = (__bf16)s1[j];
      }
      v16bf ap = *(const v16bf*)&P[n][16 * hh];

      // ---- O += P V : B-operand contiguous from Vs[d][key] ----
      v16bf bv;
      bv = *(const v16bf*)&Vs[ 0 + n][16 * hh];  o0 = wmma_bf16(ap, bv, o0);
      bv = *(const v16bf*)&Vs[16 + n][16 * hh];  o1 = wmma_bf16(ap, bv, o1);
      bv = *(const v16bf*)&Vs[32 + n][16 * hh];  o2 = wmma_bf16(ap, bv, o2);
      bv = *(const v16bf*)&Vs[48 + n][16 * hh];  o3 = wmma_bf16(ap, bv, o3);
    }
    __syncthreads();
  }

  // ---- epilogue: out[b, qrow, h*64 + d] = O / l ----
  const int bb = bh / H_;
  const int h  = bh % H_;
  #pragma unroll
  for (int j = 0; j < 8; ++j) {
    const int   r   = qBase + j + 8 * hh;
    const float inv = 1.0f / lrow[j];
    float* dst = out + ((size_t)bb * N_ + r) * E_ + h * D_;
    dst[ 0 + n] = o0[j] * inv;
    dst[16 + n] = o1[j] * inv;
    dst[32 + n] = o2[j] * inv;
    dst[48 + n] = o3[j] * inv;
  }
}

// ---------------------------------------------------------------------------
extern "C" void kernel_launch(void* const* d_in, const int* in_sizes, int n_in,
                              void* d_out, int out_size, void* d_ws, size_t ws_size,
                              hipStream_t stream) {
  const float* x     = (const float*)d_in[0];
  const float* gamma = (const float*)d_in[1];
  const float* beta  = (const float*)d_in[2];
  const float* w     = (const float*)d_in[3];
  float* out = (float*)d_out;

  char* ws = (char*)d_ws;
  __bf16* xn    = (__bf16*)ws;                                     // 8 MB
  __bf16* wb    = (__bf16*)(ws + (size_t)ROWS * E_ * 2);           // 6 MB
  __bf16* heads = (__bf16*)(ws + (size_t)ROWS * E_ * 2
                               + (size_t)F3 * E_ * 2);             // 24 MB

  ln_norm_kernel<<<ROWS, 256, 0, stream>>>(x, gamma, beta, xn);
  cvt_w_kernel<<<(F3 * E_) / (256 * 4), 256, 0, stream>>>(w, wb);
  qkv_gemm_kernel<<<(ROWS / 32) * (F3 / 64) / 8, 256, 0, stream>>>(xn, wb, heads);
  attn_kernel<<<(BH * (N_ / 16)) / 8, 256, 0, stream>>>(heads, out);
}
